// CausalSelfAttention_79611513798857
// MI455X (gfx1250) — matrix operand order repaired and checked by reading, not compiled, not run on hardware
//
#include <hip/hip_runtime.h>
#include <hip/hip_bf16.h>

// ---------------------------------------------------------------------------
// Causal self-attention forward for MI455X (gfx1250, wave32, WMMA).
//   B=2, T=2048, C=2048, H=16, D=128
// Matmuls on v_wmma_f32_16x16x32_bf16 (f32 accumulate); GEMM tiles staged with
// GLOBAL_LOAD_ASYNC_TO_LDS_B128 (ASYNCcnt) into double-buffered LDS.
// ---------------------------------------------------------------------------

typedef __attribute__((ext_vector_type(16))) __bf16 bf16x16;
typedef __attribute__((ext_vector_type(8)))  __bf16 bf16x8;
typedef __attribute__((ext_vector_type(8)))  float  f32x8;

#define SHUF16(lo, hi) __builtin_shufflevector((lo), (hi), 0,1,2,3,4,5,6,7,8,9,10,11,12,13,14,15)

static __device__ __forceinline__ f32x8 wmma_bf16(bf16x16 a, bf16x16 b, f32x8 c) {
  return __builtin_amdgcn_wmma_f32_16x16x32_bf16(false, a, false, b, (short)0, c, false, false);
}

// Async copy global -> LDS, 16B per lane, tracked by ASYNCcnt (ISA §10, op 98).
static __device__ __forceinline__ void async_copy_b128(unsigned lds_addr,
                                                       unsigned long long gaddr) {
  asm volatile("global_load_async_to_lds_b128 %0, %1, off"
               :: "v"(lds_addr), "v"(gaddr)
               : "memory");
}
static __device__ __forceinline__ unsigned lds_addr_of(const void* p) {
  return (unsigned)(unsigned long long)p;  // flat->LDS uses addr[31:0] (ISA 10.2)
}

// ---------------------------------------------------------------------------
// Kernel 1: fp32 -> bf16 elementwise convert
// ---------------------------------------------------------------------------
__global__ __launch_bounds__(256) void convert_f32_bf16_kernel(
    const float* __restrict__ src, __bf16* __restrict__ dst, long long n) {
  long long i = (long long)blockIdx.x * blockDim.x + threadIdx.x;
  long long stride = (long long)gridDim.x * blockDim.x;
  for (; i < n; i += stride) dst[i] = (__bf16)src[i];
}

// ---------------------------------------------------------------------------
// Kernel 2: fp32 [K][N] -> bf16 [N][K] (tiled transpose through LDS)
// ---------------------------------------------------------------------------
__global__ __launch_bounds__(256) void transpose_convert_kernel(
    const float* __restrict__ W, __bf16* __restrict__ WT, int K, int N) {
  __shared__ float tile[32][33];
  int k0 = blockIdx.y * 32, n0 = blockIdx.x * 32;
  int tx = threadIdx.x & 31, ty = threadIdx.x >> 5;  // ty = 0..7
#pragma unroll
  for (int i = 0; i < 32; i += 8)
    tile[ty + i][tx] = W[(size_t)(k0 + ty + i) * N + (n0 + tx)];
  __syncthreads();
#pragma unroll
  for (int i = 0; i < 32; i += 8)
    WT[(size_t)(n0 + ty + i) * K + (k0 + tx)] = (__bf16)tile[tx][ty + i];
}

// ---------------------------------------------------------------------------
// Kernel 3: bf16 WMMA GEMM.  C[M][N] = A[M][K] * BT[N][K]^T
// Block tile 256x128, 256 threads = 8 waves (4 along M x 2 along N),
// wave tile 64x64 = 4x4 WMMA 16x16 tiles, K-step 32.
// Double-buffered LDS fed by GLOBAL_LOAD_ASYNC_TO_LDS_B128:
//   issue tile i+1 into buf[1-cur], s_wait_asynccnt 6 (drains tile i, in-order),
//   barrier, 16 WMMAs from buf[cur], barrier.
// ---------------------------------------------------------------------------
#define GEMM_LDS_STRIDE 40  // bf16 elems per LDS row (80B: 16B-aligned, conflict-free)

template <typename OutT>
__global__ __launch_bounds__(256) void gemm_bf16_kernel(
    const __bf16* __restrict__ A, const __bf16* __restrict__ BT,
    OutT* __restrict__ C, int M, int N, int K) {
  __shared__ __bf16 sA[2][256 * GEMM_LDS_STRIDE];
  __shared__ __bf16 sB[2][128 * GEMM_LDS_STRIDE];

  const int tid = threadIdx.x;
  const int lane = tid & 31;
  const int wave = tid >> 5;
  const int bm = blockIdx.y * 256;
  const int bn = blockIdx.x * 128;
  const int wm = (wave & 3) * 64;   // wave M offset in block tile
  const int wn = (wave >> 2) * 64;  // wave N offset in block tile

  const int frow   = lane & 15;               // row/col within a 16-wide tile
  const int khalfA = (lane < 16) ? 0 : 8;     // A-frag k base (ISA 16-bit A layout)
  const int khalfB = (lane < 16) ? 0 : 8;     // B-frag k base (ISA 16-bit B layout)

  const f32x8 zero = {};
  f32x8 acc[4][4];
#pragma unroll
  for (int i = 0; i < 4; ++i)
#pragma unroll
    for (int j = 0; j < 4; ++j) acc[i][j] = zero;

  // Cooperative async staging: A = 256 rows x 64B (1 thread/row, 4x b128),
  //                            B = 128 rows x 64B (2 threads/row, 2x b128).
  const int brow = tid >> 1;
  const int bcol = (tid & 1) * 16;

  auto issue_tile = [&](int kk, int buf) {
    unsigned long long gA = (unsigned long long)(A + (size_t)(bm + tid) * K + kk);
    unsigned lA = lds_addr_of(&sA[buf][tid * GEMM_LDS_STRIDE]);
#pragma unroll
    for (int e = 0; e < 4; ++e)
      async_copy_b128(lA + e * 16, gA + (unsigned long long)(e * 16));
    unsigned long long gB = (unsigned long long)(BT + (size_t)(bn + brow) * K + kk + bcol);
    unsigned lB = lds_addr_of(&sB[buf][brow * GEMM_LDS_STRIDE + bcol]);
#pragma unroll
    for (int e = 0; e < 2; ++e)
      async_copy_b128(lB + e * 16, gB + (unsigned long long)(e * 16));
  };

  const int nk = K / 32;
  issue_tile(0, 0);

  for (int kt = 0; kt < nk; ++kt) {
    const int cur = kt & 1;
    if (kt + 1 < nk) {
      issue_tile((kt + 1) * 32, cur ^ 1);
      asm volatile("s_wait_asynccnt 0x6" ::: "memory");  // tile kt landed (in-order)
    } else {
      asm volatile("s_wait_asynccnt 0x0" ::: "memory");
    }
    __syncthreads();

    bf16x16 afrag[4];
#pragma unroll
    for (int i = 0; i < 4; ++i) {
      const __bf16* p = &sA[cur][(wm + i * 16 + frow) * GEMM_LDS_STRIDE];
      bf16x8 lo = *(const bf16x8*)(p + khalfA);
      bf16x8 hi = *(const bf16x8*)(p + khalfA + 16);
      afrag[i] = SHUF16(lo, hi);
    }
#pragma unroll
    for (int j = 0; j < 4; ++j) {
      const __bf16* p = &sB[cur][(wn + j * 16 + frow) * GEMM_LDS_STRIDE + khalfB];
      bf16x8 lo = *(const bf16x8*)(p);
      bf16x8 hi = *(const bf16x8*)(p + 16);
      bf16x16 bfrag = SHUF16(lo, hi);
#pragma unroll
      for (int i = 0; i < 4; ++i)
        acc[i][j] = wmma_bf16(afrag[i], bfrag, acc[i][j]);
    }
    __syncthreads();
  }

  const int g8 = (lane < 16) ? 0 : 8;  // C/D layout: element r -> row r+g8
#pragma unroll
  for (int i = 0; i < 4; ++i)
#pragma unroll
    for (int j = 0; j < 4; ++j)
#pragma unroll
      for (int r = 0; r < 8; ++r) {
        int row = bm + wm + i * 16 + r + g8;
        int col = bn + wn + j * 16 + frow;
        C[(size_t)row * N + col] = (OutT)acc[i][j][r];
      }
}

// ---------------------------------------------------------------------------
// Kernel 4: split qkv [B*T][3C] (bf16) -> RoPE'd Q/K and V in [B,H,T,D] bf16
// ---------------------------------------------------------------------------
__global__ __launch_bounds__(256) void rope_split_kernel(
    const __bf16* __restrict__ qkv, const float* __restrict__ cosT,
    const float* __restrict__ sinT, __bf16* __restrict__ Qh,
    __bf16* __restrict__ Kh, __bf16* __restrict__ Vh, int Tlen) {
  const int idx = blockIdx.x * blockDim.x + threadIdx.x;  // (b,t,h,d<64)
  const int d = idx & 63;
  const int h = (idx >> 6) & 15;
  const int bt = idx >> 10;
  const int t = bt % Tlen;
  const int b = bt / Tlen;
  if (b >= 2) return;

  const size_t row = ((size_t)b * Tlen + t) * (3 * 2048);
  const size_t qo = row + (size_t)h * 128;
  const float c = cosT[t * 64 + d];
  const float s = sinT[t * 64 + d];
  const float q1 = (float)qkv[qo + d],        q2 = (float)qkv[qo + 64 + d];
  const float k1 = (float)qkv[qo + 2048 + d], k2 = (float)qkv[qo + 2048 + 64 + d];

  const size_t hb = (((size_t)b * 16 + h) * Tlen + t) * 128;
  Qh[hb + d]      = (__bf16)(q1 * c - q2 * s);
  Qh[hb + 64 + d] = (__bf16)(q1 * s + q2 * c);
  Kh[hb + d]      = (__bf16)(k1 * c - k2 * s);
  Kh[hb + 64 + d] = (__bf16)(k1 * s + k2 * c);
  Vh[hb + d]      = qkv[qo + 4096 + d];
  Vh[hb + 64 + d] = qkv[qo + 4096 + 64 + d];
}

// ---------------------------------------------------------------------------
// Kernel 5: flash attention. One block = (b,h, 128 query rows), 8 waves x 16 rows.
// Key loop in steps of 32: S = Q*K^T (8 WMMA), online softmax (shfl_xor over
// 16-lane groups), P staged through per-wave LDS into A-frag form, O += P*V
// (8 WMMA). Output written bf16 to [B,T,H*D] for the projection GEMM.
// ---------------------------------------------------------------------------
#define SK_STRIDE 136  // 32 keys x 128 d, row stride (272B, 16B aligned)
#define SV_STRIDE 40   // 128 d x 32 keys (transposed), row stride (80B)
#define SP_STRIDE 40   // per-wave 16x32 P tile row stride

__global__ __launch_bounds__(256) void flash_attn_kernel(
    const __bf16* __restrict__ Qh, const __bf16* __restrict__ Kh,
    const __bf16* __restrict__ Vh, __bf16* __restrict__ O, int Tlen) {
  __shared__ __bf16 sK[32 * SK_STRIDE];
  __shared__ __bf16 sV[128 * SV_STRIDE];
  __shared__ __bf16 sP[8][16 * SP_STRIDE];

  const int tid = threadIdx.x;
  const int lane = tid & 31;
  const int wave = tid >> 5;
  const int bh = blockIdx.y;           // b*16 + h
  const int qt = blockIdx.x;           // 128-row query tile
  const size_t base = (size_t)bh * Tlen * 128;

  const int frow   = lane & 15;
  const int khalfA = (lane < 16) ? 0 : 8;
  const int khalfB = (lane < 16) ? 0 : 8;
  const int g8     = (lane < 16) ? 0 : 8;

  const int qrow = qt * 128 + wave * 16;  // this wave's first query row

  // Q fragments: 4 chunks of 32 along D, A-frag layout, kept in registers.
  bf16x16 qf[4];
#pragma unroll
  for (int c = 0; c < 4; ++c) {
    const __bf16* p = Qh + base + (size_t)(qrow + frow) * 128 + c * 32;
    bf16x8 lo = *(const bf16x8*)(p + khalfA);
    bf16x8 hi = *(const bf16x8*)(p + khalfA + 16);
    qf[c] = SHUF16(lo, hi);
  }

  const f32x8 zero = {};
  f32x8 oacc[8];
#pragma unroll
  for (int dt = 0; dt < 8; ++dt) oacc[dt] = zero;
  float mrow[8], lrow[8];
#pragma unroll
  for (int r = 0; r < 8; ++r) { mrow[r] = -1e30f; lrow[r] = 0.0f; }

  const float scale = 0.08838834764831845f;  // 1/sqrt(128)
  const int kend = qt * 128 + 128;           // causal bound (== multiple of 32)

  // Cooperative K/V staging addressing: 8 threads per 128-d row, 16 bf16 each.
  const int kr = tid >> 3;
  const int kc = (tid & 7) * 16;

  for (int k0 = 0; k0 < kend; k0 += 32) {
    {  // stage K row-major, V transposed (d-major) into LDS
      const __bf16* gk = Kh + base + (size_t)(k0 + kr) * 128 + kc;
      const __bf16* gv = Vh + base + (size_t)(k0 + kr) * 128 + kc;
      bf16x8 a0 = *(const bf16x8*)(gk);
      bf16x8 a1 = *(const bf16x8*)(gk + 8);
      bf16x8 v0 = *(const bf16x8*)(gv);
      bf16x8 v1 = *(const bf16x8*)(gv + 8);
      __syncthreads();
      *(bf16x8*)(&sK[kr * SK_STRIDE + kc]) = a0;
      *(bf16x8*)(&sK[kr * SK_STRIDE + kc + 8]) = a1;
#pragma unroll
      for (int e = 0; e < 8; ++e) {
        sV[(kc + e) * SV_STRIDE + kr] = v0[e];
        sV[(kc + 8 + e) * SV_STRIDE + kr] = v1[e];
      }
    }
    __syncthreads();

    // S = Q * K^T : two 16-key subtiles x 4 d-chunks
    f32x8 s[2];
#pragma unroll
    for (int n = 0; n < 2; ++n) {
      s[n] = zero;
#pragma unroll
      for (int c = 0; c < 4; ++c) {
        const __bf16* p = &sK[(n * 16 + frow) * SK_STRIDE + c * 32 + khalfB];
        bf16x8 lo = *(const bf16x8*)(p);
        bf16x8 hi = *(const bf16x8*)(p + 16);
        bf16x16 kf = SHUF16(lo, hi);
        s[n] = wmma_bf16(qf[c], kf, s[n]);
      }
    }

    // Online softmax per row (row r+g8 of this wave's 16 rows)
#pragma unroll
    for (int r = 0; r < 8; ++r) {
      const int q = qrow + r + g8;
      float v0 = s[0][r] * scale;
      float v1 = s[1][r] * scale;
      if (k0 + frow > q) v0 = -1e30f;
      if (k0 + 16 + frow > q) v1 = -1e30f;
      float vmax = fmaxf(v0, v1);
#pragma unroll
      for (int off = 8; off >= 1; off >>= 1)
        vmax = fmaxf(vmax, __shfl_xor(vmax, off, 32));  // stays in 16-lane group
      const float mnew = fmaxf(mrow[r], vmax);
      const float corr = __expf(mrow[r] - mnew);
      const float p0 = __expf(v0 - mnew);
      const float p1 = __expf(v1 - mnew);
      float psum = p0 + p1;
#pragma unroll
      for (int off = 8; off >= 1; off >>= 1)
        psum += __shfl_xor(psum, off, 32);
      lrow[r] = lrow[r] * corr + psum;
      mrow[r] = mnew;
#pragma unroll
      for (int dt = 0; dt < 8; ++dt) oacc[dt][r] *= corr;
      sP[wave][(r + g8) * SP_STRIDE + frow] = (__bf16)p0;
      sP[wave][(r + g8) * SP_STRIDE + 16 + frow] = (__bf16)p1;
    }
    __builtin_amdgcn_wave_barrier();  // wave-local LDS P transpose: order stores/loads

    // Reload P in A-frag layout (16x32)
    bf16x16 pf;
    {
      const __bf16* p = &sP[wave][frow * SP_STRIDE];
      bf16x8 lo = *(const bf16x8*)(p + khalfA);
      bf16x8 hi = *(const bf16x8*)(p + khalfA + 16);
      pf = SHUF16(lo, hi);
    }

    // O += P * V : 8 output d-tiles, V from transposed LDS tile
#pragma unroll
    for (int dt = 0; dt < 8; ++dt) {
      const __bf16* p = &sV[(dt * 16 + frow) * SV_STRIDE + khalfB];
      bf16x8 lo = *(const bf16x8*)(p);
      bf16x8 hi = *(const bf16x8*)(p + 16);
      bf16x16 vf = SHUF16(lo, hi);
      oacc[dt] = wmma_bf16(pf, vf, oacc[dt]);
    }
    __syncthreads();
  }

  // Epilogue: O /= l, write bf16 [B, T, H*D] (== [B,T,C] for projection GEMM)
  const int b = bh >> 4, h = bh & 15;
#pragma unroll
  for (int r = 0; r < 8; ++r) {
    const float inv = 1.0f / lrow[r];
    const int t = qrow + r + g8;
    const size_t orow = ((size_t)b * Tlen + t) * 2048 + (size_t)h * 128;
#pragma unroll
    for (int dt = 0; dt < 8; ++dt)
      O[orow + dt * 16 + frow] = (__bf16)(oacc[dt][r] * inv);
  }
}

// ---------------------------------------------------------------------------
// Host-side launch
// ---------------------------------------------------------------------------
extern "C" void kernel_launch(void* const* d_in, const int* in_sizes, int n_in,
                              void* d_out, int out_size, void* d_ws, size_t ws_size,
                              hipStream_t stream) {
  constexpr int Bn = 2, T = 2048, Cc = 2048, H = 16, D = 128;
  constexpr long long M = (long long)Bn * T;       // 4096
  constexpr int N3 = 3 * Cc;                       // 6144

  const float* x      = (const float*)d_in[0];
  const float* cosT   = (const float*)d_in[1];
  const float* sinT   = (const float*)d_in[2];
  const float* W_qkv  = (const float*)d_in[3];
  const float* W_proj = (const float*)d_in[4];
  float* out = (float*)d_out;

  // Workspace layout (all 256B aligned)
  char* ws = (char*)d_ws;
  size_t off = 0;
  auto alloc = [&](size_t bytes) { void* p = ws + off; off = (off + bytes + 255) & ~(size_t)255; return p; };
  __bf16* x_bf   = (__bf16*)alloc(M * Cc * 2);           // [M][C]
  __bf16* WqkvT  = (__bf16*)alloc((size_t)N3 * Cc * 2);  // [3C][C]
  __bf16* WprojT = (__bf16*)alloc((size_t)Cc * Cc * 2);  // [C][C]
  __bf16* qkv_bf = (__bf16*)alloc(M * N3 * 2);           // [M][3C]
  __bf16* Qh     = (__bf16*)alloc(M * Cc * 2);           // [B,H,T,D]
  __bf16* Kh     = (__bf16*)alloc(M * Cc * 2);
  __bf16* Vh     = (__bf16*)alloc(M * Cc * 2);
  __bf16* attnO  = (__bf16*)alloc(M * Cc * 2);           // [B,T,C]

  // 1) x -> bf16
  convert_f32_bf16_kernel<<<4096, 256, 0, stream>>>(x, x_bf, M * Cc);
  // 2) weight transpose+convert: [K][N] fp32 -> [N][K] bf16
  transpose_convert_kernel<<<dim3(N3 / 32, Cc / 32), 256, 0, stream>>>(W_qkv, WqkvT, Cc, N3);
  transpose_convert_kernel<<<dim3(Cc / 32, Cc / 32), 256, 0, stream>>>(W_proj, WprojT, Cc, Cc);
  // 3) qkv = x @ W_qkv  (bf16 out)
  gemm_bf16_kernel<__bf16><<<dim3(N3 / 128, M / 256), 256, 0, stream>>>(
      x_bf, WqkvT, qkv_bf, (int)M, N3, Cc);
  // 4) RoPE + head split
  rope_split_kernel<<<(Bn * T * H * (D / 2)) / 256, 256, 0, stream>>>(
      qkv_bf, cosT, sinT, Qh, Kh, Vh, T);
  // 5) flash attention
  flash_attn_kernel<<<dim3(T / 128, Bn * H), 256, 0, stream>>>(Qh, Kh, Vh, attnO, T);
  // 6) out = attnO @ W_proj  (fp32 out)
  gemm_bf16_kernel<float><<<dim3(Cc / 128, M / 256), 256, 0, stream>>>(
      attnO, WprojT, out, (int)M, Cc, Cc);
  (void)in_sizes; (void)n_in; (void)out_size; (void)ws_size;
}
